// Backbone_70600672411800
// MI455X (gfx1250) — matrix operand order repaired and checked by reading, not compiled, not run on hardware
//
#include <hip/hip_runtime.h>
#include <cstdint>
#include <cstddef>

// ---------------------------------------------------------------------------
// Types / WMMA helpers (CDNA5 gfx1250, wave32)
// ---------------------------------------------------------------------------
typedef _Float16 h16;
typedef __attribute__((ext_vector_type(16))) _Float16 v16h;
typedef __attribute__((ext_vector_type(8)))  float    v8f;
typedef __attribute__((ext_vector_type(4))) unsigned int u32x4;
typedef __attribute__((ext_vector_type(4))) int          i32x4;
typedef __attribute__((ext_vector_type(8))) int          i32x8;

#define DEVINL __device__ __forceinline__

// A fragment: 16x32 f16 tile, row-major source with leading dim `ld`.
// Layout (ISA 7.12.2): lane L (L<16) holds row M=L, K = k0+[0..7] and k0+16+[0..7];
// lane L+16 holds row M=L, K = k0+8+[0..7] and k0+24+[0..7].
DEVINL v16h load_a_frag(const h16* __restrict__ base, int ld, int k0) {
  int lane = threadIdx.x & 31;
  int row  = lane & 15;
  int half = lane >> 4;
  const h16* p = base + (size_t)row * ld + k0 + half * 8;
  v16h a;
#pragma unroll
  for (int t = 0; t < 8; ++t) a[t] = p[t];
#pragma unroll
  for (int t = 0; t < 8; ++t) a[8 + t] = p[16 + t];
  return a;
}

// B fragment: 32x16 f16 tile from weight stored (K,N) row-major (leading dim ldn).
// Lane L holds column N=col0+L%16; K = k0+8h+[0..7] and k0+16+8h+[0..7] (h=L/16).
DEVINL v16h load_b_frag(const h16* __restrict__ Wb, int ldn, int k0, int col0) {
  int lane = threadIdx.x & 31;
  int half = lane >> 4;
  const h16* p = Wb + (size_t)(k0 + half * 8) * ldn + col0 + (lane & 15);
  v16h b;
#pragma unroll
  for (int t = 0; t < 8; ++t) b[t] = p[(size_t)t * ldn];
#pragma unroll
  for (int t = 0; t < 8; ++t) b[8 + t] = p[(size_t)(16 + t) * ldn];
  return b;
}

DEVINL v8f wmma_f16(v16h a, v16h b, v8f c) {
  return __builtin_amdgcn_wmma_f32_16x16x32_f16(false, a, false, b, (short)0, c,
                                                false, false);
}

// CDNA5 async global->LDS copy (ASYNCcnt); per-lane 16B (ISA 15.18.3 op 98).
DEVINL void async_g2l_b128(unsigned int lds_off, const void* gaddr) {
  asm volatile("global_load_async_to_lds_b128 %0, %1, off"
               :
               : "v"(lds_off), "v"(gaddr)
               : "memory");
}
DEVINL void s_wait_async0() {
  asm volatile("s_wait_asynccnt 0x0" ::: "memory");
}

// ---------------------------------------------------------------------------
// Generic WMMA GEMM:  out = epilog( A(M,K)f16 @ Wb(K,N)f16 )  (wave per 16 x 16*NT strip)
// epilog: (+bias)*scale, optional relu, optional +resid, write f32 and/or f16.
// Requires M%16==0, K%32==0, N in {32,64,multiples of 128 chunks}.
// ---------------------------------------------------------------------------
template <int NT>
__global__ void __launch_bounds__(128) gemm_kernel(
    const h16* __restrict__ A, const h16* __restrict__ Wb,
    const float* __restrict__ bias, const float* __restrict__ resid,
    float* __restrict__ outf, h16* __restrict__ outh,
    int M, int N, int K, float scale, int do_relu) {
  int wave = threadIdx.x >> 5;
  int lane = threadIdx.x & 31;
  int rowbase = (blockIdx.x * 4 + wave) * 16;
  if (rowbase >= M) return;
  int colchunk = blockIdx.y * 128;

  v8f acc[NT];
#pragma unroll
  for (int t = 0; t < NT; ++t) {
#pragma unroll
    for (int q = 0; q < 8; ++q) acc[t][q] = 0.0f;
  }

  const h16* Arow = A + (size_t)rowbase * K;
  __builtin_prefetch(Arow + (size_t)(lane & 15) * K, 0, 0);
  for (int k0 = 0; k0 < K; k0 += 32) {
    v16h afr = load_a_frag(Arow, K, k0);
#pragma unroll
    for (int t = 0; t < NT; ++t) {
      v16h bfr = load_b_frag(Wb, N, k0, colchunk + t * 16);
      acc[t] = wmma_f16(afr, bfr, acc[t]);
    }
  }

  int half = lane >> 4, cl = lane & 15;
#pragma unroll
  for (int t = 0; t < NT; ++t) {
    int col = colchunk + t * 16 + cl;
#pragma unroll
    for (int r = 0; r < 8; ++r) {
      int row = rowbase + r + 8 * half;
      float v = acc[t][r];
      if (bias) v += bias[col];
      v *= scale;
      if (do_relu) v = fmaxf(v, 0.0f);
      size_t o = (size_t)row * N + col;
      if (resid) v += resid[o];
      if (outf) outf[o] = v;
      if (outh) outh[o] = (h16)v;
    }
  }
}

static void gemmF(hipStream_t st, const h16* A, const h16* Wb, const float* bias,
                  const float* resid, float* outf, h16* outh,
                  int M, int N, int K, float scale, int relu) {
  dim3 blk(128);
  dim3 grd((unsigned)((M + 63) / 64), (unsigned)((N + 127) / 128));
  if ((N & 127) == 0)
    gemm_kernel<8><<<grd, blk, 0, st>>>(A, Wb, bias, resid, outf, outh, M, N, K, scale, relu);
  else if (N == 64)
    gemm_kernel<4><<<grd, blk, 0, st>>>(A, Wb, bias, resid, outf, outh, M, N, K, scale, relu);
  else if (N == 32)
    gemm_kernel<2><<<grd, blk, 0, st>>>(A, Wb, bias, resid, outf, outh, M, N, K, scale, relu);
  else
    gemm_kernel<1><<<grd, blk, 0, st>>>(A, Wb, bias, resid, outf, outh, M, N, K, scale, relu);
}

// ---------------------------------------------------------------------------
// Specialized K=128,N=128 projection GEMM (q/k/v): the 32KB weight panel is
// staged into LDS once per block by the Tensor Data Mover (one 2-D descriptor,
// issued by wave 0, TENSORcnt-tracked), then shared by all 4 waves.
// D# packing per ISA sections 8.3/8.4: group0 = {count=1 | lds_addr |
// global_addr | type=2}; group1 = {data_size=2B, tensor_dim0/1=128,
// tile_dim0/1=128, tensor_dim0_stride=128}; groups 2/3 zero (2-D tile).
// ---------------------------------------------------------------------------
__global__ void __launch_bounds__(128) gemm128_tdm_kernel(
    const h16* __restrict__ A, const h16* __restrict__ Wb,
    float* __restrict__ outf, int M) {
  __shared__ h16 sW[128 * 128];
  int wave = threadIdx.x >> 5;
  int lane = threadIdx.x & 31;

  if (wave == 0) {
    uint64_t ga = (uint64_t)(uintptr_t)Wb;
    unsigned int lds = (unsigned int)(uintptr_t)(&sW[0]);  // low 32 bits = LDS offset
    u32x4 g0;
    g0[0] = 1u;                                            // count=1 (valid user D#)
    g0[1] = lds;                                           // lds_addr (bytes)
    g0[2] = (unsigned int)ga;                              // global_addr[31:0]
    g0[3] = (unsigned int)((ga >> 32) & 0x01FFFFFFu) | (2u << 30);  // addr[56:32] | type=2
    i32x8 g1;
    g1[0] = (int)(1u << 16);    // workgroup_mask=0, data_size=1 (2 bytes)
    g1[1] = (int)(128u << 16);  // tensor_dim0[15:0]=128 in bits[79:64->31:16]
    g1[2] = (int)(128u << 16);  // tensor_dim0 hi=0 | tensor_dim1[15:0]=128
    g1[3] = (int)(128u << 16);  // tensor_dim1 hi=0 | tile_dim0=128
    g1[4] = 128;                // tile_dim1=128, tile_dim2=0
    g1[5] = 128;                // tensor_dim0_stride=128 (elements)
    g1[6] = 0;                  // stride hi / tensor_dim1_stride lo
    g1[7] = 0;                  // tensor_dim1_stride hi
    i32x4 gz4;
    gz4[0] = gz4[1] = gz4[2] = gz4[3] = 0;
    i32x8 gz8;
#pragma unroll
    for (int t = 0; t < 8; ++t) gz8[t] = 0;
    // amdgpu-toolchain (clang-23) 6-arg form: (g0, g1, g2, g3, extra, cpol)
    __builtin_amdgcn_tensor_load_to_lds(g0, g1, gz4, gz4, gz8, 0);
    __builtin_amdgcn_s_wait_tensorcnt(0);
  }
  __syncthreads();

  int rowbase = (blockIdx.x * 4 + wave) * 16;  // M is always a multiple of 64 here
  v8f acc[8];
#pragma unroll
  for (int t = 0; t < 8; ++t) {
#pragma unroll
    for (int q = 0; q < 8; ++q) acc[t][q] = 0.0f;
  }
  const h16* Arow = A + (size_t)rowbase * 128;
#pragma unroll
  for (int k0 = 0; k0 < 128; k0 += 32) {
    v16h afr = load_a_frag(Arow, 128, k0);
#pragma unroll
    for (int t = 0; t < 8; ++t) {
      v16h bfr = load_b_frag(sW, 128, k0, t * 16);
      acc[t] = wmma_f16(afr, bfr, acc[t]);
    }
  }
  int half = lane >> 4, cl = lane & 15;
#pragma unroll
  for (int t = 0; t < 8; ++t) {
#pragma unroll
    for (int r = 0; r < 8; ++r) {
      int row = rowbase + r + 8 * half;
      outf[(size_t)row * 128 + t * 16 + cl] = acc[t][r];
    }
  }
}

// ---------------------------------------------------------------------------
// Fused point-transformer attention: one block (8 wave32) per query point.
// pos = relu(dxyz@d1+b)@d2+b ; attn_in = q - k_gather + pos ;
// logits = relu(attn_in@g1+b)@g2+b ; softmax over neighbors; res = sum p*(v+pos)
// The 16 gathered k/v rows are brought into LDS with async global->LDS copies
// issued as soon as the kNN indices are known, overlapping the delta-MLP WMMA.
// ---------------------------------------------------------------------------
__global__ void __launch_bounds__(256) attn_kernel(
    const float* __restrict__ xyz, const int* __restrict__ knn,
    const float* __restrict__ qf, const float* __restrict__ kf,
    const float* __restrict__ vf,
    const float* __restrict__ d1w, const float* __restrict__ d1b,
    const h16* __restrict__ d2h, const float* __restrict__ d2b,
    const h16* __restrict__ g1h, const float* __restrict__ g1b,
    const h16* __restrict__ g2h, const float* __restrict__ g2b,
    h16* __restrict__ resh, int n, int kk) {
  __shared__ h16   sA[16 * 128];
  __shared__ h16   sA2[16 * 128];
  __shared__ float sPos[16 * 128];
  __shared__ float sLog[16 * 128];
  __shared__ float sK[16 * 128];
  __shared__ float sV[16 * 128];
  __shared__ float sQ[128];
  __shared__ float sDx[48];
  __shared__ int   sIdx[16];

  int tid = threadIdx.x;
  int wave = tid >> 5, lane = tid & 31;
  int b = blockIdx.x / n;
  int i = blockIdx.x - b * n;
  size_t base = (size_t)b * n + i;

  if (tid < 16) sIdx[tid] = (tid < kk) ? knn[base * 16 + tid] : 0;
  if (tid < 128) sQ[tid] = qf[base * 128 + tid];
  __syncthreads();

  // Issue async gathers of the 16 k rows and 16 v rows (8KB each) into LDS.
  {
    unsigned int kbase = (unsigned int)(uintptr_t)(&sK[0]);
    unsigned int vbase = (unsigned int)(uintptr_t)(&sV[0]);
#pragma unroll
    for (int cc = 0; cc < 2; ++cc) {
      int e16 = tid + cc * 256;          // 16-byte chunk id (512 chunks = 8KB)
      int j = e16 >> 5;                  // 32 chunks per 512B row
      int off = (e16 & 31) * 16;         // byte offset within row
      const char* gk = (const char*)(kf + ((size_t)b * n + sIdx[j]) * 128) + off;
      const char* gv = (const char*)(vf + ((size_t)b * n + sIdx[j]) * 128) + off;
      async_g2l_b128(kbase + (unsigned int)(j * 512 + off), gk);
      async_g2l_b128(vbase + (unsigned int)(j * 512 + off), gv);
    }
  }

  if (tid < 48) {
    int j = tid / 3, c = tid - j * 3;
    sDx[tid] = xyz[base * 3 + c] - xyz[((size_t)b * n + sIdx[j]) * 3 + c];
  }
  __syncthreads();

  // h1 = relu(dxyz @ delta1 + b) -> sA (zero for j >= kk)
  for (int e = tid; e < 2048; e += 256) {
    int j = e >> 7, c = e & 127;
    float v = 0.0f;
    if (j < kk) {
      v = d1b[c] + sDx[j * 3 + 0] * d1w[c] + sDx[j * 3 + 1] * d1w[128 + c] +
          sDx[j * 3 + 2] * d1w[256 + c];
      v = fmaxf(v, 0.0f);
    }
    sA[e] = (h16)v;
  }
  __syncthreads();

  int half = lane >> 4, cl = lane & 15, col0 = wave * 16;

  // pos = h1 @ delta2 + b  -> sPos (f32)
  {
    v8f acc;
#pragma unroll
    for (int q = 0; q < 8; ++q) acc[q] = 0.0f;
#pragma unroll
    for (int k0 = 0; k0 < 128; k0 += 32) {
      v16h afr = load_a_frag(sA, 128, k0);
      v16h bfr = load_b_frag(d2h, 128, k0, col0);
      acc = wmma_f16(afr, bfr, acc);
    }
#pragma unroll
    for (int r = 0; r < 8; ++r) {
      int row = r + 8 * half;
      sPos[row * 128 + col0 + cl] = acc[r] + d2b[col0 + cl];
    }
  }
  s_wait_async0();   // all waves: own async gathers complete
  __syncthreads();   // ... and everyone else's

  // attn_in = q - k_gather + pos -> sA (f16), zero for j >= kk
  for (int e = tid; e < 2048; e += 256) {
    int j = e >> 7, c = e & 127;
    float v = 0.0f;
    if (j < kk)
      v = sQ[c] - sK[j * 128 + c] + sPos[j * 128 + c];
    sA[e] = (h16)v;
  }
  __syncthreads();

  // h2 = relu(attn_in @ gamma1 + b) -> sA2 (f16)
  {
    v8f acc;
#pragma unroll
    for (int q = 0; q < 8; ++q) acc[q] = 0.0f;
#pragma unroll
    for (int k0 = 0; k0 < 128; k0 += 32) {
      v16h afr = load_a_frag(sA, 128, k0);
      v16h bfr = load_b_frag(g1h, 128, k0, col0);
      acc = wmma_f16(afr, bfr, acc);
    }
#pragma unroll
    for (int r = 0; r < 8; ++r) {
      int row = r + 8 * half;
      int col = col0 + cl;
      sA2[row * 128 + col] = (h16)fmaxf(acc[r] + g1b[col], 0.0f);
    }
  }
  __syncthreads();

  // logits = h2 @ gamma2 + b -> sLog (f32)
  {
    v8f acc;
#pragma unroll
    for (int q = 0; q < 8; ++q) acc[q] = 0.0f;
#pragma unroll
    for (int k0 = 0; k0 < 128; k0 += 32) {
      v16h afr = load_a_frag(sA2, 128, k0);
      v16h bfr = load_b_frag(g2h, 128, k0, col0);
      acc = wmma_f16(afr, bfr, acc);
    }
#pragma unroll
    for (int r = 0; r < 8; ++r) {
      int row = r + 8 * half;
      int col = col0 + cl;
      sLog[row * 128 + col] = acc[r] + g2b[col];
    }
  }
  __syncthreads();

  // per-channel softmax over neighbors + weighted sum of (v + pos)
  if (tid < 128) {
    int c = tid;
    const float sc = 0.08838834764831845f;  // 1/sqrt(128)
    float m = -3.0e38f;
    for (int j = 0; j < kk; ++j) m = fmaxf(m, sLog[j * 128 + c] * sc);
    float s = 0.0f, r = 0.0f;
    for (int j = 0; j < kk; ++j) {
      float p = expf(sLog[j * 128 + c] * sc - m);
      s += p;
      r += p * (sV[j * 128 + c] + sPos[j * 128 + c]);
    }
    resh[base * 128 + c] = (h16)(r / s);
  }
}

// ---------------------------------------------------------------------------
// Elementwise / support kernels
// ---------------------------------------------------------------------------
__global__ void feats_kernel(const float* __restrict__ x,
                             const float* __restrict__ w1, const float* __restrict__ b1,
                             const float* __restrict__ w2, const float* __restrict__ b2,
                             float* __restrict__ xyz, float* __restrict__ ff,
                             h16* __restrict__ fh, int total) {
  int p = blockIdx.x * blockDim.x + threadIdx.x;
  if (p >= total) return;
  float in[5];
#pragma unroll
  for (int t = 0; t < 5; ++t) in[t] = x[(size_t)p * 5 + t];
#pragma unroll
  for (int c = 0; c < 3; ++c) xyz[(size_t)p * 3 + c] = in[c];
  float h[32];
#pragma unroll
  for (int o = 0; o < 32; ++o) {
    float v = b1[o];
#pragma unroll
    for (int t = 0; t < 5; ++t) v += in[t] * w1[t * 32 + o];
    h[o] = fmaxf(v, 0.0f);
  }
  for (int o = 0; o < 32; ++o) {
    float v = b2[o];
    for (int t = 0; t < 32; ++t) v += h[t] * w2[t * 32 + o];
    ff[(size_t)p * 32 + o] = v;
    fh[(size_t)p * 32 + o] = (h16)v;
  }
}

__global__ void convert_w_kernel(const float* __restrict__ W, h16* __restrict__ Wh,
                                 int K, int N, int Kpad) {
  int e = blockIdx.x * blockDim.x + threadIdx.x;
  int tot = Kpad * N;
  if (e >= tot) return;
  int k = e / N, c = e - k * N;
  Wh[e] = (k < K) ? (h16)W[(size_t)k * N + c] : (h16)0.0f;
}

// register top-16 nearest neighbors, first-index tie-break (matches lax.top_k)
__global__ void knn_kernel(const float* __restrict__ q, const float* __restrict__ s,
                           int* __restrict__ outidx, int nq, int ns, int B_) {
  int g = blockIdx.x * blockDim.x + threadIdx.x;
  if (g >= B_ * nq) return;
  int b = g / nq;
  float qx = q[(size_t)g * 3], qy = q[(size_t)g * 3 + 1], qz = q[(size_t)g * 3 + 2];
  float bd[16];
  int bi[16];
#pragma unroll
  for (int t = 0; t < 16; ++t) { bd[t] = 3.0e38f; bi[t] = 0; }
  const float* sp = s + (size_t)b * ns * 3;
  for (int j = 0; j < ns; ++j) {
    float dx = sp[(size_t)j * 3] - qx;
    float dy = sp[(size_t)j * 3 + 1] - qy;
    float dz = sp[(size_t)j * 3 + 2] - qz;
    float d = dx * dx + dy * dy + dz * dz;
#pragma unroll
    for (int t = 15; t >= 1; --t) {
      if (d < bd[t - 1])      { bd[t] = bd[t - 1]; bi[t] = bi[t - 1]; }
      else if (d < bd[t])     { bd[t] = d;          bi[t] = j; }
    }
    if (d < bd[0]) { bd[0] = d; bi[0] = j; }
  }
#pragma unroll
  for (int t = 0; t < 16; ++t) outidx[(size_t)g * 16 + t] = bi[t];
}

// farthest point sampling, one block per batch, exact reference semantics
__global__ void __launch_bounds__(256) fps_kernel(const float* __restrict__ xyz,
                                                  int n, int npoint,
                                                  int* __restrict__ out) {
  __shared__ float sD[2048];
  __shared__ float sV[256];
  __shared__ int sI[256];
  __shared__ int sFar;
  int b = blockIdx.x, tid = threadIdx.x;
  const float* xp = xyz + (size_t)b * n * 3;
  for (int p = tid; p < n; p += 256) sD[p] = 1.0e10f;
  if (tid == 0) sFar = 0;
  __syncthreads();
  for (int it = 0; it < npoint; ++it) {
    int far = sFar;
    if (tid == 0) out[b * npoint + it] = far;
    float cx = xp[(size_t)far * 3], cy = xp[(size_t)far * 3 + 1], cz = xp[(size_t)far * 3 + 2];
    float bv = -1.0f;
    int bi = 0;
    for (int p = tid; p < n; p += 256) {
      float dx = xp[(size_t)p * 3] - cx;
      float dy = xp[(size_t)p * 3 + 1] - cy;
      float dz = xp[(size_t)p * 3 + 2] - cz;
      float d = dx * dx + dy * dy + dz * dz;
      float nd = fminf(sD[p], d);
      sD[p] = nd;
      if (nd > bv) { bv = nd; bi = p; }
    }
    sV[tid] = bv; sI[tid] = bi;
    __syncthreads();
    for (int off = 128; off > 0; off >>= 1) {
      if (tid < off) {
        float ov = sV[tid + off]; int oi = sI[tid + off];
        if (ov > sV[tid] || (ov == sV[tid] && oi < sI[tid])) { sV[tid] = ov; sI[tid] = oi; }
      }
      __syncthreads();
    }
    if (tid == 0) sFar = sI[0];
    __syncthreads();
  }
}

__global__ void gather_xyz_kernel(const float* __restrict__ src, const int* __restrict__ idx,
                                  float* __restrict__ dst, int nq, int ns, int B_) {
  int e = blockIdx.x * blockDim.x + threadIdx.x;
  if (e >= B_ * nq * 3) return;
  int c = e % 3;
  int g = e / 3;
  int b = g / nq;
  int sidx = idx[g];
  dst[e] = src[((size_t)b * ns + sidx) * 3 + c];
}

// build grouped features [dxyz(3) | gathered points(Cprev) | zero pad] as f16
__global__ void group_kernel(const float* __restrict__ sxyz, const float* __restrict__ nxyz,
                             const int* __restrict__ idx, const float* __restrict__ pts,
                             h16* __restrict__ G, int nq, int ns, int Cprev, int Kpad) {
  size_t e = (size_t)blockIdx.x * blockDim.x + threadIdx.x;
  size_t tot = (size_t)16 * nq * 16 * Kpad;
  if (e >= tot) return;
  int c = (int)(e % Kpad);
  size_t row = e / Kpad;
  int j = (int)(row % 16);
  size_t bq = row / 16;  // b*nq + q
  int b = (int)(bq / nq);
  int s = idx[bq * 16 + j];
  float v;
  if (c < 3)
    v = sxyz[((size_t)b * ns + s) * 3 + c] - nxyz[bq * 3 + c];
  else if (c < 3 + Cprev)
    v = pts[((size_t)b * ns + s) * Cprev + (c - 3)];
  else
    v = 0.0f;
  G[e] = (h16)v;
}

__global__ void maxpool_kernel(const float* __restrict__ g2, float* __restrict__ outf,
                               h16* __restrict__ outh, int nq, int ch, int B_) {
  int e = blockIdx.x * blockDim.x + threadIdx.x;
  if (e >= B_ * nq * ch) return;
  int c = e % ch;
  int g = e / ch;  // b*nq+q
  const float* p = g2 + ((size_t)g * 16) * ch + c;
  float m = -3.0e38f;
  for (int j = 0; j < 16; ++j) m = fmaxf(m, p[(size_t)j * ch]);
  outf[e] = m;
  outh[e] = (h16)m;
}

// selective scan: 512 independent (batch,state) recurrences, writes h sequence (f16)
__global__ void scan_kernel(const float* __restrict__ xb, const float* __restrict__ Ap,
                            h16* __restrict__ hseq, int n, int B_) {
  int t = blockIdx.x * blockDim.x + threadIdx.x;
  if (t >= B_ * 32) return;
  int b = t >> 5, s = t & 31;
  float a = -expf(Ap[s]);
  float h = 0.0f;
  const float* xp = xb + (size_t)b * n * 32 + s;
  h16* hp = hseq + (size_t)b * n * 32 + s;
  for (int i = 0; i < n; ++i) {
    float xv = xp[(size_t)i * 32];
    float d = (xv > 20.0f) ? xv : log1pf(expf(xv));  // softplus
    h = h * expf(d * a) + xv;
    hp[(size_t)i * 32] = (h16)h;
  }
}

__global__ void scan_add_kernel(float* __restrict__ pts, h16* __restrict__ ptsh,
                                const float* __restrict__ y, const float* __restrict__ D,
                                int total, int ch) {
  int e = blockIdx.x * blockDim.x + threadIdx.x;
  if (e >= total) return;
  int c = e % ch;
  float p = pts[e];
  float v = p + y[e] + p * D[c];
  pts[e] = v;
  ptsh[e] = (h16)v;
}

__global__ void mean_kernel(const float* __restrict__ pts, float* __restrict__ out,
                            int n, int ch, int B_) {
  int e = blockIdx.x * blockDim.x + threadIdx.x;
  if (e >= B_ * ch) return;
  int b = e / ch, c = e - b * ch;
  float s = 0.0f;
  for (int t = 0; t < n; ++t) s += pts[((size_t)b * n + t) * ch + c];
  out[e] = s / (float)n;
}

// ---------------------------------------------------------------------------
// Host-side parameter mapping & orchestration
// ---------------------------------------------------------------------------
struct Lin { const float *w, *b; };
struct TBp { Lin fc1, fc2, d1, d2, g1, g2; const float *wq, *wk, *wv; };
struct Blk { Lin td0, td1; TBp tf; const float *A, *Bm, *Cm, *D; };
struct Net { const float* x; Lin fc1a, fc1b; TBp t0; Blk blk[4]; };

static const float* FP(void* const* in, int& k) { return (const float*)in[k++]; }
static Lin LN(void* const* in, int& k) { Lin l; l.w = FP(in, k); l.b = FP(in, k); return l; }

static void tb_ins(TBp& t, void* const* in, int& k) {
  t.fc1 = LN(in, k); t.fc2 = LN(in, k); t.d1 = LN(in, k); t.d2 = LN(in, k);
  t.g1 = LN(in, k); t.g2 = LN(in, k);
  t.wq = FP(in, k); t.wk = FP(in, k); t.wv = FP(in, k);
}
static void tb_sorted(TBp& t, void* const* in, int& k) {
  t.d1 = LN(in, k); t.d2 = LN(in, k); t.fc1 = LN(in, k); t.fc2 = LN(in, k);
  t.g1 = LN(in, k); t.g2 = LN(in, k);
  t.wk = FP(in, k); t.wq = FP(in, k); t.wv = FP(in, k);
}
static void blk_ins(Blk& b, void* const* in, int& k) {
  b.td0 = LN(in, k); b.td1 = LN(in, k);
  tb_ins(b.tf, in, k);
  b.A = FP(in, k); b.Bm = FP(in, k); b.Cm = FP(in, k); b.D = FP(in, k);
}
static void blk_sorted(Blk& b, void* const* in, int& k) {
  b.A = FP(in, k); b.Bm = FP(in, k); b.Cm = FP(in, k); b.D = FP(in, k);
  b.td0 = LN(in, k); b.td1 = LN(in, k);
  tb_sorted(b.tf, in, k);
}

static void fill_net(Net& P, void* const* in, const int* sizes) {
  int k = 0;
  if (sizes[0] == 16 * 2048 * 5) {
    P.x = FP(in, k);
    if (sizes[1] == 160) {  // full insertion order
      P.fc1a = LN(in, k); P.fc1b = LN(in, k);
      tb_ins(P.t0, in, k);
      for (int i = 0; i < 4; ++i) blk_ins(P.blk[i], in, k);
    } else {                // x first, params in sorted tree-leaf order
      for (int i = 0; i < 4; ++i) blk_sorted(P.blk[i], in, k);
      P.fc1a = LN(in, k); P.fc1b = LN(in, k);
      tb_sorted(P.t0, in, k);
    }
  } else {                  // fully sorted tree-leaf order, x last
    for (int i = 0; i < 4; ++i) blk_sorted(P.blk[i], in, k);
    P.fc1a = LN(in, k); P.fc1b = LN(in, k);
    tb_sorted(P.t0, in, k);
    P.x = FP(in, k);
  }
}

struct Arena {
  char* base; size_t off; size_t cap;
  void* take(size_t bytes) {
    size_t a = (off + 255) & ~(size_t)255;
    off = a + bytes;
    return base + a;
  }
};

struct TBh { const h16 *fc1, *fc2, *d2, *g1, *g2, *wq, *wk, *wv; };

static const h16* convw(hipStream_t st, Arena& ar, const float* W, int K, int N, int Kpad) {
  h16* d = (h16*)ar.take((size_t)Kpad * N * sizeof(h16));
  int tot = Kpad * N;
  convert_w_kernel<<<(tot + 255) / 256, 256, 0, st>>>(W, d, K, N, Kpad);
  return d;
}

static TBh conv_tb(hipStream_t st, Arena& ar, const TBp& p, int dpts) {
  TBh h;
  h.fc1 = convw(st, ar, p.fc1.w, dpts, 128, dpts);
  h.fc2 = convw(st, ar, p.fc2.w, 128, dpts, 128);
  h.d2  = convw(st, ar, p.d2.w, 128, 128, 128);
  h.g1  = convw(st, ar, p.g1.w, 128, 128, 128);
  h.g2  = convw(st, ar, p.g2.w, 128, 128, 128);
  h.wq  = convw(st, ar, p.wq, 128, 128, 128);
  h.wk  = convw(st, ar, p.wk, 128, 128, 128);
  h.wv  = convw(st, ar, p.wv, 128, 128, 128);
  return h;
}

static void run_tblock(hipStream_t st, const TBp& p, const TBh& ph,
                       const float* xyz, int n, int dpts, int kk,
                       const float* inF, const h16* inH,
                       float* outF, h16* outH,
                       int* knnbuf, h16* xh, float* qf, float* kf, float* vf, h16* resh) {
  int M = 16 * n;
  knn_kernel<<<(M + 127) / 128, 128, 0, st>>>(xyz, xyz, knnbuf, n, n, 16);
  gemmF(st, inH, ph.fc1, p.fc1.b, nullptr, nullptr, xh, M, 128, dpts, 1.0f, 0);
  {
    dim3 g((unsigned)(M / 64)), b2(128);
    gemm128_tdm_kernel<<<g, b2, 0, st>>>(xh, ph.wq, qf, M);
    gemm128_tdm_kernel<<<g, b2, 0, st>>>(xh, ph.wk, kf, M);
    gemm128_tdm_kernel<<<g, b2, 0, st>>>(xh, ph.wv, vf, M);
  }
  attn_kernel<<<M, 256, 0, st>>>(xyz, knnbuf, qf, kf, vf, p.d1.w, p.d1.b,
                                 ph.d2, p.d2.b, ph.g1, p.g1.b, ph.g2, p.g2.b,
                                 resh, n, kk);
  gemmF(st, resh, ph.fc2, p.fc2.b, inF, outF, outH, M, dpts, 128, 1.0f, 0);
}

extern "C" void kernel_launch(void* const* d_in, const int* in_sizes, int n_in,
                              void* d_out, int out_size, void* d_ws, size_t ws_size,
                              hipStream_t stream) {
  (void)out_size;
  if (n_in < 112) return;
  if (ws_size < (size_t)170 * 1024 * 1024) return;  // workspace layout needs ~155 MB

  Net P;
  fill_net(P, d_in, in_sizes);
  Arena ar{(char*)d_ws, 0, ws_size};

  const int B = 16, N0 = 2048;
  const int CHs[4] = {64, 128, 256, 512};

  // activation workspace
  float* xyzA = (float*)ar.take((size_t)B * N0 * 3 * 4);
  float* xyzB = (float*)ar.take((size_t)B * N0 * 3 * 4);
  float* ptsF[2]; h16* ptsH[2];
  for (int t = 0; t < 2; ++t) {
    ptsF[t] = (float*)ar.take((size_t)B * 65536 * 4);
    ptsH[t] = (h16*)ar.take((size_t)B * 65536 * 2);
  }
  h16*   xh   = (h16*)ar.take((size_t)B * N0 * 128 * 2);
  float* qf   = (float*)ar.take((size_t)B * N0 * 128 * 4);
  float* kf   = (float*)ar.take((size_t)B * N0 * 128 * 4);
  float* vf   = (float*)ar.take((size_t)B * N0 * 128 * 4);
  h16*   resh = (h16*)ar.take((size_t)B * N0 * 128 * 2);
  int*   knnb = (int*)ar.take((size_t)B * N0 * 16 * 4);
  int*   fpsb = (int*)ar.take((size_t)B * 512 * 4);
  h16*   Gb   = (h16*)ar.take((size_t)8388608 * 2);
  h16*   g1h  = (h16*)ar.take((size_t)8388608 * 2);
  float* g2f  = (float*)ar.take((size_t)8388608 * 4);
  float* xbf  = (float*)ar.take((size_t)262144 * 4);
  h16*   hsh  = (h16*)ar.take((size_t)262144 * 2);
  float* yf   = (float*)ar.take((size_t)524288 * 4);

  // weight conversions (f32 -> f16, (K,N) layout == CDNA5 B-fragment layout)
  TBh t0h = conv_tb(stream, ar, P.t0, 32);
  TBh tfh[4];
  const h16 *tdW1h[4], *tdW2h[4], *Bmh[4], *Cmh[4];
  for (int i = 0; i < 4; ++i) {
    int ch = CHs[i];
    int Kin = ch / 2 + 3;
    int Kpad = (Kin + 31) & ~31;
    tdW1h[i] = convw(stream, ar, P.blk[i].td0.w, Kin, ch, Kpad);
    tdW2h[i] = convw(stream, ar, P.blk[i].td1.w, ch, ch, ch);
    tfh[i] = conv_tb(stream, ar, P.blk[i].tf, ch);
    Bmh[i] = convw(stream, ar, P.blk[i].Bm, ch, 32, ch);
    Cmh[i] = convw(stream, ar, P.blk[i].Cm, 32, ch, 32);
  }

  // stage 0: xyz split + 5->32->32 MLP
  {
    int tot = B * N0;
    feats_kernel<<<(tot + 255) / 256, 256, 0, stream>>>(
        P.x, P.fc1a.w, P.fc1a.b, P.fc1b.w, P.fc1b.b, xyzA, ptsF[0], ptsH[0], tot);
  }
  run_tblock(stream, P.t0, t0h, xyzA, N0, 32, 16, ptsF[0], ptsH[0], ptsF[1], ptsH[1],
             knnb, xh, qf, kf, vf, resh);

  float* curXyz = xyzA;
  float* othXyz = xyzB;
  int cur = 1;
  int n = N0;
  const float bn_s = 0.999995000037f;  // 1/sqrt(1 + 1e-5)

  for (int i = 0; i < 4; ++i) {
    int np = N0 >> (2 * (i + 1));
    int ch = CHs[i];
    int Cprev = ch / 2;
    int Kin = Cprev + 3;
    int Kpad = (Kin + 31) & ~31;

    // transition down
    fps_kernel<<<B, 256, 0, stream>>>(curXyz, n, np, fpsb);
    {
      int tot = B * np * 3;
      gather_xyz_kernel<<<(tot + 255) / 256, 256, 0, stream>>>(curXyz, fpsb, othXyz, np, n, B);
    }
    {
      int tot = B * np;
      knn_kernel<<<(tot + 127) / 128, 128, 0, stream>>>(othXyz, curXyz, knnb, np, n, B);
    }
    {
      size_t tot = (size_t)B * np * 16 * Kpad;
      group_kernel<<<(unsigned)((tot + 255) / 256), 256, 0, stream>>>(
          curXyz, othXyz, knnb, ptsF[cur], Gb, np, n, Cprev, Kpad);
    }
    gemmF(stream, Gb, tdW1h[i], P.blk[i].td0.b, nullptr, nullptr, g1h,
          B * np * 16, ch, Kpad, bn_s, 1);
    gemmF(stream, g1h, tdW2h[i], P.blk[i].td1.b, nullptr, g2f, nullptr,
          B * np * 16, ch, ch, bn_s, 1);
    {
      int tot = B * np * ch;
      maxpool_kernel<<<(tot + 255) / 256, 256, 0, stream>>>(g2f, ptsF[1 - cur],
                                                            ptsH[1 - cur], np, ch, B);
    }
    { float* t = curXyz; curXyz = othXyz; othXyz = t; }

    // transformer block at the new resolution
    int kk = np < 16 ? np : 16;
    run_tblock(stream, P.blk[i].tf, tfh[i], curXyz, np, ch, kk,
               ptsF[1 - cur], ptsH[1 - cur], ptsF[cur], ptsH[cur],
               knnb, xh, qf, kf, vf, resh);

    // selective scan + residual
    gemmF(stream, ptsH[cur], Bmh[i], nullptr, nullptr, xbf, nullptr, B * np, 32, ch, 1.0f, 0);
    scan_kernel<<<2, 256, 0, stream>>>(xbf, P.blk[i].A, hsh, np, B);
    gemmF(stream, hsh, Cmh[i], nullptr, nullptr, yf, nullptr, B * np, ch, 32, 1.0f, 0);
    {
      int tot = B * np * ch;
      scan_add_kernel<<<(tot + 255) / 256, 256, 0, stream>>>(ptsF[cur], ptsH[cur], yf,
                                                             P.blk[i].D, tot, ch);
    }
    n = np;
  }

  mean_kernel<<<(B * 512 + 255) / 256, 256, 0, stream>>>(ptsF[cur], (float*)d_out, 8, 512, B);
}
